// MCNET_28991029248225
// MI455X (gfx1250) — 8-bit matrix operand order repaired and checked by reading, compiled, not run on hardware
//
#include <hip/hip_runtime.h>
#include <hip/hip_bf16.h>
#include <stdint.h>

typedef __attribute__((ext_vector_type(8))) int v8i;


__device__ __forceinline__ float fsign(float v) {
    return v > 0.f ? 1.f : (v < 0.f ? -1.f : 0.f);
}
__device__ __forceinline__ int8_t isign(int s) {
    return s > 0 ? (int8_t)1 : (s < 0 ? (int8_t)-1 : (int8_t)0);
}

__global__ __launch_bounds__(256) void sign_weights_kernel(
    const float* __restrict__ w, int8_t* __restrict__ wb,
    int cout, int cin, int kpad, int coutp) {
    int idx = blockIdx.x * 256 + threadIdx.x;
    int total = coutp * kpad;
    if (idx >= total) return;
    int co = idx / kpad;
    int k  = idx - co * kpad;
    int keff = cin * 9;
    int8_t s = 0;
    if (co < cout && k < keff) {
        int q  = k / cin;
        int ci = k - q * cin;
        float v = w[(size_t)co * keff + ci * 9 + q];
        s = v > 0.f ? (int8_t)1 : (v < 0.f ? (int8_t)-1 : (int8_t)0);
    }
    wb[(size_t)co * kpad + k] = s;
}

__global__ __launch_bounds__(256) void conv0_pool_sign_kernel(
    const float* __restrict__ x, const float* __restrict__ w0,
    int8_t* __restrict__ act0) {
    __shared__ float sw[108];
    int tid = threadIdx.x;
    if (tid < 108) sw[tid] = fsign(w0[tid]);
    __syncthreads();

    int idx = blockIdx.x * 256 + tid;
    if (idx >= 64 * 127 * 127) return;
    int px = idx % 127; int t = idx / 127;
    int py = t % 127;   int n = t / 127;

    float xp[3][4][4];
    #pragma unroll
    for (int ci = 0; ci < 3; ++ci)
        #pragma unroll
        for (int r = 0; r < 4; ++r)
            #pragma unroll
            for (int c = 0; c < 4; ++c)
                xp[ci][r][c] =
                    x[(((size_t)n * 3 + ci) * 256 + (2 * py + r)) * 256 + (2 * px + c)];

    #pragma unroll
    for (int co = 0; co < 4; ++co) {
        const float* wsl = &sw[co * 27];
        float best = -3.4e38f;
        #pragma unroll
        for (int dy = 0; dy < 2; ++dy)
            #pragma unroll
            for (int dx = 0; dx < 2; ++dx) {
                float acc = 0.f;
                #pragma unroll
                for (int ci = 0; ci < 3; ++ci)
                    #pragma unroll
                    for (int kh = 0; kh < 3; ++kh)
                        #pragma unroll
                        for (int kw = 0; kw < 3; ++kw)
                            acc += xp[ci][dy + kh][dx + kw] * wsl[ci * 9 + kh * 3 + kw];
                best = fmaxf(best, acc);
            }
        act0[(((size_t)n * 4 + co) * 127 + py) * 127 + px] =
            best > 0.f ? (int8_t)1 : (best < 0.f ? (int8_t)-1 : (int8_t)0);
    }
}

template <int CIN, int COUT, int H, int W, bool FINAL>
__global__ __launch_bounds__(2 * (((COUT + 15) / 16) * 16)) void bconv_kernel(
    const int8_t* __restrict__ in, const int8_t* __restrict__ wb,
    int8_t* __restrict__ out8, float* __restrict__ outf) {
    constexpr int OUTH   = H - 2;
    constexpr int OUTW   = W - 2;
    constexpr int KEFF   = CIN * 9;
    constexpr int KPAD   = ((KEFF + 63) / 64) * 64;
    constexpr int NCHUNK = KPAD / 64;
    constexpr int TILESX = (OUTW + 15) / 16;
    constexpr int COUTP  = ((COUT + 15) / 16) * 16;
    constexpr int NT     = COUTP / 16;
    constexpr int NTHR   = 32 * NT;
    constexpr int ZOFF   = CIN * 54;
    static_assert((CIN & 3) == 0, "CIN multiple of 4 for word loads");

    const int tid   = threadIdx.x;
    const int ntile = tid >> 5;
    const int lane  = tid & 31;
    const int hl    = lane >> 4;
    const int l16   = lane & 15;

    const int oy  = blockIdx.x / TILESX;
    const int ox0 = (blockIdx.x % TILESX) * 16;
    const int n   = blockIdx.z;

    __shared__ __align__(16) unsigned char patch[CIN * 54 + 4];
    for (int t = tid; t < CIN * 54; t += NTHR) {
        int ci = t / 54, r = t - ci * 54;
        int kh = r / 18, col = r - kh * 18;
        int gx = ox0 + col; if (gx > W - 1) gx = W - 1;
        patch[(kh * 18 + col) * CIN + ci] = (unsigned char)
            in[(((size_t)n * CIN + ci) * H + (oy + kh)) * W + gx];
    }
    if (tid < 4) patch[ZOFF + tid] = 0;
    __syncthreads();

    const int cob = ntile * 16 + l16;
    const int8_t* wrow = wb + (size_t)cob * KPAD;
    __builtin_prefetch(wrow, 0, 1);

    v8i cacc = {};
    #pragma unroll
    for (int ch = 0; ch < NCHUNK; ++ch) {
        const int koff = ch * 64;
        v8i af;
        #pragma unroll
        for (int v = 0; v < 8; ++v) {
            const int k0  = koff + hl * 8 + (v & 1) * 4 + ((v >> 1) << 4);
            const int q   = k0 / CIN;
            const int ci0 = k0 % CIN;
            const int f   = ((q / 3) * 18 + (q % 3)) * CIN + ci0;
            const int adr = (k0 < KEFF) ? (l16 * CIN + f) : ZOFF;
            af[v] = *(const int*)(patch + adr);
        }
        v8i bf;
        #pragma unroll
        for (int v = 0; v < 8; ++v) {
            const int k0 = koff + hl * 8 + (v & 1) * 4 + ((v >> 1) << 4);
            bf[v] = *(const int*)(wrow + k0);
        }
        cacc = __builtin_amdgcn_wmma_i32_16x16x64_iu8(
             true, af,  true, bf, cacc,
             false,  false);
    }

    const int co = ntile * 16 + l16;
    const size_t obase = ((size_t)n * COUT + co) * OUTH + oy;
    if (ox0 + 16 <= OUTW && co < COUT) {
        #pragma unroll
        for (int j = 0; j < 8; ++j) {
            const int ox = ox0 + j + 8 * hl;
            const int s  = cacc[j];
            if constexpr (FINAL)
                outf[obase * OUTW + ox] = s > 1 ? 1.f : (s < -1 ? -1.f : (float)s);
            else
                out8[obase * OUTW + ox] = isign(s);
        }
    } else {
        #pragma unroll
        for (int j = 0; j < 8; ++j) {
            const int ox = ox0 + j + 8 * hl;
            if (ox < OUTW && co < COUT) {
                const int s = cacc[j];
                if constexpr (FINAL)
                    outf[obase * OUTW + ox] = s > 1 ? 1.f : (s < -1 ? -1.f : (float)s);
                else
                    out8[obase * OUTW + ox] = isign(s);
            }
        }
    }
}

template <int CIN, int COUT, int H, int W, bool FINAL>
static void launch_bconv(const int8_t* in, const int8_t* wb, int8_t* out8,
                         float* outf, hipStream_t stream) {
    constexpr int OUTH = H - 2, OUTW = W - 2;
    constexpr int TILESX = (OUTW + 15) / 16;
    constexpr int COUTP = ((COUT + 15) / 16) * 16;
    dim3 grid(OUTH * TILESX, 1, 64);
    bconv_kernel<CIN, COUT, H, W, FINAL>
        <<<grid, 32 * (COUTP / 16), 0, stream>>>(in, wb, out8, outf);
}

static void launch_signw(const float* w, int8_t* wb, int cout, int cin,
                         hipStream_t stream) {
    int keff = cin * 9;
    int kpad = ((keff + 63) / 64) * 64;
    int coutp = ((cout + 15) / 16) * 16;
    int total = coutp * kpad;
    sign_weights_kernel<<<(total + 255) / 256, 256, 0, stream>>>(
        w, wb, cout, cin, kpad, coutp);
}

extern "C" void kernel_launch(void* const* d_in, const int* in_sizes, int n_in,
                              void* d_out, int out_size, void* d_ws, size_t ws_size,
                              hipStream_t stream) {
    const float* x  = (const float*)d_in[0];
    const float* w0 = (const float*)d_in[1];
    const float* w[6] = {(const float*)d_in[2], (const float*)d_in[3],
                         (const float*)d_in[4], (const float*)d_in[5],
                         (const float*)d_in[6], (const float*)d_in[7]};
    float* out = (float*)d_out;

    size_t off = 0;
    auto carve = [&](size_t bytes) -> int8_t* {
        off = (off + 255) & ~(size_t)255;
        int8_t* p = (int8_t*)d_ws + off;
        off += bytes;
        return p;
    };
    int8_t* act0 = carve((size_t)64 * 4  * 127 * 127);
    int8_t* act1 = carve((size_t)64 * 8  * 125 * 125);
    int8_t* act2 = carve((size_t)64 * 16 * 123 * 123);
    int8_t* act3 = carve((size_t)64 * 32 * 121 * 121);
    int8_t* act4 = carve((size_t)64 * 64 * 119 * 119);
    int8_t* act5 = carve((size_t)64 * 32 * 117 * 117);
    int8_t* wb1 = carve(16 * 64);
    int8_t* wb2 = carve(16 * 128);
    int8_t* wb3 = carve(32 * 192);
    int8_t* wb4 = carve(64 * 320);
    int8_t* wb5 = carve(32 * 576);
    int8_t* wb6 = carve(16 * 320);
    if (off > ws_size) return;

    launch_signw(w[0], wb1, 8,  4,  stream);
    launch_signw(w[1], wb2, 16, 8,  stream);
    launch_signw(w[2], wb3, 32, 16, stream);
    launch_signw(w[3], wb4, 64, 32, stream);
    launch_signw(w[4], wb5, 32, 64, stream);
    launch_signw(w[5], wb6, 2,  32, stream);

    conv0_pool_sign_kernel<<<(64 * 127 * 127 + 255) / 256, 256, 0, stream>>>(
        x, w0, act0);

    launch_bconv<4,  8,  127, 127, false>(act0, wb1, act1, nullptr, stream);
    launch_bconv<8,  16, 125, 125, false>(act1, wb2, act2, nullptr, stream);
    launch_bconv<16, 32, 123, 123, false>(act2, wb3, act3, nullptr, stream);
    launch_bconv<32, 64, 121, 121, false>(act3, wb4, act4, nullptr, stream);
    launch_bconv<64, 32, 119, 119, false>(act4, wb5, act5, nullptr, stream);
    launch_bconv<32, 2,  117, 117, true >(act5, wb6, nullptr, out, stream);
}
